// DependencyGCN_66511863546172
// MI455X (gfx1250) — compile-verified, matrix operand order checked by reading, not run on hardware
//
#include <hip/hip_runtime.h>
#include <hip/hip_bf16.h>

// ---------------------------------------------------------------------------
// Types for CDNA5 WMMA (wave32)
// ---------------------------------------------------------------------------
typedef __attribute__((ext_vector_type(16))) __bf16 v16bf;
typedef __attribute__((ext_vector_type(8)))  __bf16 v8bf;
typedef __attribute__((ext_vector_type(8)))  float  v8f;

// async-to-LDS builtin takes int4-vector pointers (AS1 global, AS3 shared)
typedef int v4i __attribute__((vector_size(16)));
typedef __attribute__((address_space(1))) v4i* as1_v4i_ptr;
typedef __attribute__((address_space(3))) v4i* as3_v4i_ptr;

#define D_HID 256

#if __has_builtin(__builtin_amdgcn_global_load_async_to_lds_b128)
#define HAVE_ASYNC_LDS 1
#else
#define HAVE_ASYNC_LDS 0
#endif

__device__ __forceinline__ void wait_async0() {
#if __has_builtin(__builtin_amdgcn_s_wait_asynccnt)
    __builtin_amdgcn_s_wait_asynccnt(0);
#else
    asm volatile("s_wait_asynccnt 0x0" ::: "memory");
#endif
}

__device__ __forceinline__ unsigned short f2bf(float f) {
    unsigned u = __float_as_uint(f);
    unsigned r = u + 0x7FFFu + ((u >> 16) & 1u);   // round-to-nearest-even
    return (unsigned short)(r >> 16);
}

// ---------------------------------------------------------------------------
// Degree / normalization prep
// ---------------------------------------------------------------------------
__global__ void deg_init_kernel(float* dis, int n) {
    int i = blockIdx.x * blockDim.x + threadIdx.x;
    if (i < n) dis[i] = 1.0f;                      // self-loop weight
}

__global__ void deg_accum_kernel(const long long* __restrict__ dst,
                                 const float* __restrict__ w,
                                 float* dis, int e) {
    int i = blockIdx.x * blockDim.x + threadIdx.x;
    if (i < e) atomicAdd(&dis[dst[i]], w[i]);
}

__global__ void norm_node_kernel(float* dis, float* normSelf, int n) {
    int i = blockIdx.x * blockDim.x + threadIdx.x;
    if (i < n) {
        float d = dis[i];
        float r = (d > 0.0f) ? rsqrtf(d) : 0.0f;
        dis[i] = r;
        normSelf[i] = r * r;                        // self-loop norm (w=1)
    }
}

__global__ void norm_edge_kernel(const long long* __restrict__ src,
                                 const long long* __restrict__ dst,
                                 const float* __restrict__ w,
                                 const float* __restrict__ dis,
                                 float* normE, int e) {
    int i = blockIdx.x * blockDim.x + threadIdx.x;
    if (i < e) normE[i] = dis[src[i]] * w[i] * dis[dst[i]];
}

// ---------------------------------------------------------------------------
// fp32 -> bf16 conversion of activations
// ---------------------------------------------------------------------------
__global__ void cvt_bf16_kernel(const float* __restrict__ x,
                                unsigned short* __restrict__ out, size_t n) {
    size_t i = (size_t)blockIdx.x * blockDim.x + threadIdx.x;
    if (i < n) out[i] = f2bf(x[i]);
}

// ---------------------------------------------------------------------------
// Pack W [256(K) x 256(N)] fp32 row-major into the exact per-lane B-fragment
// layout for v_wmma_f32_16x16x32_bf16:
//   Bp[nt(16)][kb(8)][lane(32)][e(16)]  (bf16)
//   lane: col = nt*16 + (lane&15), half = lane>>4
//   e<8 : K = kb*32 + half*8 + e
//   e>=8: K = kb*32 + 16 + half*8 + (e-8)
// ---------------------------------------------------------------------------
__global__ void pack_w_kernel(const float* __restrict__ W,
                              unsigned short* __restrict__ Bp) {
    int t = blockIdx.x * blockDim.x + threadIdx.x;   // 65536 threads
    int e    = t & 15;
    int lane = (t >> 4) & 31;
    int kb   = (t >> 9) & 7;
    int nt   = t >> 12;
    int col  = nt * 16 + (lane & 15);
    int half = lane >> 4;
    int k    = kb * 32 + ((e < 8) ? (half * 8 + e) : (16 + half * 8 + (e - 8)));
    Bp[t] = f2bf(W[k * D_HID + col]);
}

// ---------------------------------------------------------------------------
// GEMM: C[M,256] = A[M,256](bf16) @ Wpacked, fp32 accumulate via WMMA.
// 256 threads = 8 waves; block owns a 32-row stripe (2 M-tiles).
// Wave w: M-tile = w>>2, N-tiles = (w&3)*4 .. +3  -> each A fragment feeds
// 4 WMMAs; 32 WMMAs per wave, 256 per block.
// A stripe (16 KB) staged via GLOBAL_LOAD_ASYNC_TO_LDS_B128 when available.
// ---------------------------------------------------------------------------
__global__ __launch_bounds__(256)
void gemm_bf16_wmma_kernel(const __bf16* __restrict__ A,
                           const __bf16* __restrict__ Bp,
                           float* __restrict__ C, int M) {
    __shared__ __bf16 As[32 * D_HID];                // 16 KB
    const int m0 = blockIdx.x * 32;

    // ---- stage A stripe: 32 rows x 256 cols bf16 = 1024 x 16B chunks ----
    {
        const char* gbase = (const char*)(A + (size_t)m0 * D_HID);
        char*       lbase = (char*)As;
#if HAVE_ASYNC_LDS
#pragma unroll
        for (int i = 0; i < 4; ++i) {
            size_t off = (size_t)(threadIdx.x + i * 256) * 16;
            __builtin_amdgcn_global_load_async_to_lds_b128(
                (as1_v4i_ptr)(gbase + off),
                (as3_v4i_ptr)(lbase + off), 0, 0);
        }
        wait_async0();
#else
        const uint4* Ag   = (const uint4*)gbase;
        uint4*       As4  = (uint4*)lbase;
#pragma unroll
        for (int i = 0; i < 4; ++i) {
            int idx = threadIdx.x + i * 256;
            As4[idx] = Ag[idx];
        }
#endif
    }
    __syncthreads();

    const int wave = threadIdx.x >> 5;               // 0..7
    const int lane = threadIdx.x & 31;
    const int row  = lane & 15;
    const int half = lane >> 4;
    const int mt   = wave >> 2;                      // 0..1
    const int ng   = (wave & 3) * 4;                 // first of 4 N-tiles

    v8f acc[4] = {};
    const __bf16* Asrow = As + (mt * 16 + row) * D_HID + half * 8;

#pragma unroll
    for (int kb = 0; kb < 8; ++kb) {
        // A fragment: two 16B chunks from LDS -> v16bf
        v8bf lo = *(const v8bf*)(Asrow + kb * 32);
        v8bf hi = *(const v8bf*)(Asrow + kb * 32 + 16);
        v16bf a;
#pragma unroll
        for (int i = 0; i < 8; ++i) { a[i] = lo[i]; a[i + 8] = hi[i]; }

#pragma unroll
        for (int j = 0; j < 4; ++j) {
            // B fragment: contiguous 32B per lane from packed W (L2-hot)
            const v16bf b = *(const v16bf*)(Bp + (((ng + j) * 8 + kb) * 32 + lane) * 16);
            acc[j] = __builtin_amdgcn_wmma_f32_16x16x32_bf16(
                         false, a, false, b, (short)0, acc[j], false, false);
        }
    }

    // C/D layout: lane -> col = lane&15 ; VGPR v -> row = half*8 + v
    const int col = lane & 15;
#pragma unroll
    for (int v = 0; v < 8; ++v) {
        int r = m0 + mt * 16 + half * 8 + v;
        if (r < M) {
            float* crow = C + (size_t)r * D_HID;
#pragma unroll
            for (int j = 0; j < 4; ++j)
                crow[(ng + j) * 16 + col] = acc[j][v];
        }
    }
}

// ---------------------------------------------------------------------------
// Aggregation: G = normSelf (.) H  then  G[dst] += normE * H[src]
// ---------------------------------------------------------------------------
__global__ void self_init_kernel(const float* __restrict__ normSelf,
                                 const float* __restrict__ H,
                                 float* __restrict__ G, size_t total) {
    size_t i = (size_t)blockIdx.x * blockDim.x + threadIdx.x;
    if (i < total) G[i] = normSelf[i >> 8] * H[i];
}

__global__ __launch_bounds__(256)
void edge_agg_kernel(const long long* __restrict__ src,
                     const long long* __restrict__ dst,
                     const float* __restrict__ normE,
                     const float* __restrict__ H,
                     float* __restrict__ G) {
    const int e = blockIdx.x;                        // one edge per block
    const int f = threadIdx.x;                       // 256 features
    const long long s = src[e];
    const long long d = dst[e];
    const float w = normE[e];
    atomicAdd(&G[(size_t)d * D_HID + f], w * H[(size_t)s * D_HID + f]);
}

// bias + relu in place (fp32) and emit bf16 copy for the next GEMM
__global__ void bias_relu_cvt_kernel(float* __restrict__ G,
                                     const float* __restrict__ b,
                                     unsigned short* __restrict__ Abf,
                                     size_t total) {
    size_t i = (size_t)blockIdx.x * blockDim.x + threadIdx.x;
    if (i < total) {
        float v = G[i] + b[i & (D_HID - 1)];
        v = fmaxf(v, 0.0f);
        G[i] = v;
        Abf[i] = f2bf(v);
    }
}

// ---------------------------------------------------------------------------
// Layer 3: 256-wide dot per node (wave32 reduce), scalar aggregation, sigmoid
// ---------------------------------------------------------------------------
__global__ __launch_bounds__(256)
void w3_dot_kernel(const float* __restrict__ G, const float* __restrict__ W3,
                   float* __restrict__ h3, int n) {
    int node = blockIdx.x * 8 + (threadIdx.x >> 5);
    int lane = threadIdx.x & 31;
    if (node >= n) return;
    const float* g = G + (size_t)node * D_HID;
    float s = 0.0f;
#pragma unroll
    for (int i = 0; i < 8; ++i) s += g[lane + i * 32] * W3[lane + i * 32];
#pragma unroll
    for (int off = 16; off > 0; off >>= 1) s += __shfl_down(s, off, 32);
    if (lane == 0) h3[node] = s;
}

__global__ void agg3_init_kernel(const float* __restrict__ normSelf,
                                 const float* __restrict__ h3,
                                 float* __restrict__ z, int n) {
    int i = blockIdx.x * blockDim.x + threadIdx.x;
    if (i < n) z[i] = normSelf[i] * h3[i];
}

__global__ void agg3_edge_kernel(const long long* __restrict__ src,
                                 const long long* __restrict__ dst,
                                 const float* __restrict__ normE,
                                 const float* __restrict__ h3,
                                 float* __restrict__ z, int e) {
    int i = blockIdx.x * blockDim.x + threadIdx.x;
    if (i < e) atomicAdd(&z[dst[i]], normE[i] * h3[src[i]]);
}

__global__ void sigmoid_kernel(const float* __restrict__ z,
                               const float* __restrict__ b3,
                               float* __restrict__ out, int n) {
    int i = blockIdx.x * blockDim.x + threadIdx.x;
    if (i < n) out[i] = 1.0f / (1.0f + __expf(-(z[i] + b3[0])));
}

// ---------------------------------------------------------------------------
// Host-side orchestration
// ---------------------------------------------------------------------------
extern "C" void kernel_launch(void* const* d_in, const int* in_sizes, int n_in,
                              void* d_out, int out_size, void* d_ws, size_t ws_size,
                              hipStream_t stream) {
    const int N = in_sizes[0] / D_HID;
    const int E = in_sizes[2];

    const float*     x  = (const float*)d_in[0];
    const long long* ei = (const long long*)d_in[1];   // int64 [2,E]
    const float*     ew = (const float*)d_in[2];
    const float*     W1 = (const float*)d_in[3];
    const float*     b1 = (const float*)d_in[4];
    const float*     W2 = (const float*)d_in[5];
    const float*     b2 = (const float*)d_in[6];
    const float*     W3 = (const float*)d_in[7];
    const float*     b3 = (const float*)d_in[8];

    const long long* src = ei;
    const long long* dst = ei + E;

    // carve workspace (256B aligned chunks)
    char* p = (char*)d_ws;
    auto carve = [&](size_t bytes) {
        char* r = p;
        p += (bytes + 255) & ~(size_t)255;
        return (void*)r;
    };
    float*          dis      = (float*)carve((size_t)N * 4);
    float*          normSelf = (float*)carve((size_t)N * 4);
    float*          normE    = (float*)carve((size_t)E * 4);
    unsigned short* Abf      = (unsigned short*)carve((size_t)N * D_HID * 2);
    unsigned short* Wp       = (unsigned short*)carve((size_t)D_HID * D_HID * 2);
    float*          H        = (float*)carve((size_t)N * D_HID * 4);
    float*          G        = (float*)carve((size_t)N * D_HID * 4);
    float*          h3       = (float*)carve((size_t)N * 4);
    float*          z        = (float*)carve((size_t)N * 4);

    const size_t totalNF = (size_t)N * D_HID;
    const int nb_n    = (N + 255) / 256;
    const int nb_e    = (E + 255) / 256;
    const int nb_nf   = (int)((totalNF + 255) / 256);
    const int nb_gemm = (N + 31) / 32;

    // --- normalization prep ---
    deg_init_kernel<<<nb_n, 256, 0, stream>>>(dis, N);
    deg_accum_kernel<<<nb_e, 256, 0, stream>>>(dst, ew, dis, E);
    norm_node_kernel<<<nb_n, 256, 0, stream>>>(dis, normSelf, N);
    norm_edge_kernel<<<nb_e, 256, 0, stream>>>(src, dst, ew, dis, normE, E);

    // --- layer 1 ---
    cvt_bf16_kernel<<<nb_nf, 256, 0, stream>>>(x, Abf, totalNF);
    pack_w_kernel<<<256, 256, 0, stream>>>(W1, Wp);
    gemm_bf16_wmma_kernel<<<nb_gemm, 256, 0, stream>>>((const __bf16*)Abf, (const __bf16*)Wp, H, N);
    self_init_kernel<<<nb_nf, 256, 0, stream>>>(normSelf, H, G, totalNF);
    edge_agg_kernel<<<E, 256, 0, stream>>>(src, dst, normE, H, G);
    bias_relu_cvt_kernel<<<nb_nf, 256, 0, stream>>>(G, b1, Abf, totalNF);

    // --- layer 2 ---
    pack_w_kernel<<<256, 256, 0, stream>>>(W2, Wp);
    gemm_bf16_wmma_kernel<<<nb_gemm, 256, 0, stream>>>((const __bf16*)Abf, (const __bf16*)Wp, H, N);
    self_init_kernel<<<nb_nf, 256, 0, stream>>>(normSelf, H, G, totalNF);
    edge_agg_kernel<<<E, 256, 0, stream>>>(src, dst, normE, H, G);
    bias_relu_cvt_kernel<<<nb_nf, 256, 0, stream>>>(G, b2, Abf, totalNF);

    // --- layer 3 (output dim 1) ---
    w3_dot_kernel<<<(N + 7) / 8, 256, 0, stream>>>(G, W3, h3, N);
    agg3_init_kernel<<<nb_n, 256, 0, stream>>>(normSelf, h3, z, N);
    agg3_edge_kernel<<<nb_e, 256, 0, stream>>>(src, dst, normE, h3, z, E);
    sigmoid_kernel<<<nb_n, 256, 0, stream>>>(z, b3, (float*)d_out, N);
}
